// Individual_66288525247274
// MI455X (gfx1250) — matrix operand order repaired and checked by reading, not compiled, not run on hardware
//
#include <hip/hip_runtime.h>

#define INPUT_DIM   256
#define OUTPUT_DIM  16
#define NUM_MIDDLE  784            // M
#define NUM_NODES   1040           // N = INPUT_DIM + NUM_MIDDLE
#define BATCH       32768
#define NPAD        1040           // effT row length (k dimension), mult of 16
#define ROWSTRIDE   1048           // LDS row stride in bf16 elems (2096 B, conflict-free)
#define WAVES_PER_BLOCK 8
#define ROWS_PER_WAVE   16
#define ROWS_PER_BLOCK  (WAVES_PER_BLOCK * ROWS_PER_WAVE)   // 128

typedef __attribute__((ext_vector_type(16))) __bf16 v16bf;
typedef __attribute__((ext_vector_type(8)))  float  v8f;

__device__ __forceinline__ unsigned short f2bf(float f) {
    unsigned u = __builtin_bit_cast(unsigned, f);
    u += 0x7FFFu + ((u >> 16) & 1u);          // round-to-nearest-even
    return (unsigned short)(u >> 16);
}
__device__ __forceinline__ float bf2f(unsigned bits16) {
    return __builtin_bit_cast(float, bits16 << 16);
}
__device__ __forceinline__ float sigmoidf(float x) {
    return 1.0f / (1.0f + __expf(-x));
}

// ---------------------------------------------------------------------------
// Pre-pass: effT[i][k] = bf16( weight[k][i] * conn[k][i] * (k < 256+i) )
// Bakes the triangular "growing slice" mask into the weights, transposed so
// that a node's input column is contiguous (coalesced B-fragment loads).
// ---------------------------------------------------------------------------
__global__ void eff_transpose_kernel(const float* __restrict__ w,
                                     const int* __restrict__ conn,
                                     unsigned short* __restrict__ effT) {
    int idx = blockIdx.x * blockDim.x + threadIdx.x;
    if (idx >= NUM_MIDDLE * NUM_NODES) return;
    int k = idx / NUM_MIDDLE;       // source node row 0..1039 (coalesced reads)
    int i = idx % NUM_MIDDLE;       // middle node   0..783
    float v = 0.0f;
    if (k < INPUT_DIM + i) {
        v = w[(size_t)k * NUM_MIDDLE + i] * (float)conn[(size_t)k * NUM_MIDDLE + i];
    }
    effT[(size_t)i * NPAD + k] = f2bf(v);
}

// ---------------------------------------------------------------------------
// Main scan kernel: one wave == 16 batch rows. History tile [16 x 1040] bf16
// in LDS. 49 node-blocks of 16; each block = bf16 WMMA GEMM over the history
// + 16-step triangular fixup via ds_bpermute broadcasts.
// ---------------------------------------------------------------------------
__global__ __launch_bounds__(256) void mlp_scan_kernel(
    const float* __restrict__ x,
    const unsigned short* __restrict__ effT,
    const float* __restrict__ bias,
    const int* __restrict__ ex,
    float* __restrict__ out)
{
    extern __shared__ unsigned char lds_raw[];

    const int lane = threadIdx.x & 31;
    const int wave = threadIdx.x >> 5;
    const int n    = lane & 15;        // A-row / C-col / B-col id
    const int h    = lane >> 4;        // half-wave select
    const int hi16 = lane & 16;        // 0 or 16

    unsigned short* res = (unsigned short*)lds_raw + (size_t)wave * (ROWS_PER_WAVE * ROWSTRIDE);
    const int rowBase = blockIdx.x * ROWS_PER_BLOCK + wave * ROWS_PER_WAVE;

    // ---- 1) stage x tile: lane covers row n, columns [h*128, h*128+128) ----
    {
        const float* xr = x + (size_t)(rowBase + n) * INPUT_DIM + h * 128;
        unsigned short* dst = res + (size_t)n * ROWSTRIDE + h * 128;
        #pragma unroll 4
        for (int c = 0; c < 128; c += 4) {
            float4 f = *(const float4*)(xr + c);
            uint2 p;
            p.x = (unsigned)f2bf(f.x) | ((unsigned)f2bf(f.y) << 16);
            p.y = (unsigned)f2bf(f.z) | ((unsigned)f2bf(f.w) << 16);
            *(uint2*)(dst + c) = p;
        }
    }

    // ---- 2) zero the middle-node columns 256..1039 (784 cols = 98 x b128/row)
    {
        uint4 z; z.x = z.y = z.z = z.w = 0u;
        for (int t = lane; t < ROWS_PER_WAVE * 98; t += 32) {
            int r = t / 98, cc = t % 98;
            *(uint4*)(res + (size_t)r * ROWSTRIDE + INPUT_DIM + cc * 8) = z;
        }
    }

    union Frag { struct { uint4 lo, hi; } q; v16bf v; };

    float val[8] = {0.f, 0.f, 0.f, 0.f, 0.f, 0.f, 0.f, 0.f};

    // ---- 3) blocked sequential scan over middle nodes ----
    for (int i0 = 0; i0 < NUM_MIDDLE; i0 += 16) {
        v8f c;
        #pragma unroll
        for (int v = 0; v < 8; ++v) c[v] = 0.0f;

        const int Ktot = INPUT_DIM + i0;
        const unsigned short* bcol = effT + (size_t)(i0 + n) * NPAD + h * 16;
        const unsigned short* arow = res + (size_t)n * ROWSTRIDE + h * 16;

        if (i0 + 16 < NUM_MIDDLE)   // warm next block's weight column (global_prefetch)
            __builtin_prefetch(effT + (size_t)(i0 + 16 + n) * NPAD, 0, 0);

        // ---- GEMM over all already-computed history (K chunks of 32) ----
        for (int k0 = 0; k0 < Ktot; k0 += 32) {
            Frag a, b;
            a.q.lo = *(const uint4*)(arow + k0);         // A: K = k0+16h .. +7
            a.q.hi = *(const uint4*)(arow + k0 + 8);     //    K = k0+16h+8 .. +15
            b.q.lo = *(const uint4*)(bcol + k0);         // B: K = k0+16h .. +7
            b.q.hi = *(const uint4*)(bcol + k0 + 8);     //    K = k0+16h+8 .. +15
            c = __builtin_amdgcn_wmma_f32_16x16x32_bf16(
                    false, a.v, false, b.v, (short)0, c, false, false);
        }

        // ---- per-lane intra-block weights w[256+i0+k, i0+n], k=0..15 ----
        union { uint4 q[2]; unsigned u[8]; } wu;
        {
            const unsigned short* wp = effT + (size_t)(i0 + n) * NPAD + (INPUT_DIM + i0);
            wu.q[0] = *(const uint4*)(wp);
            wu.q[1] = *(const uint4*)(wp + 8);
        }
        float wk[16];
        #pragma unroll
        for (int k = 0; k < 16; ++k) {
            unsigned word = wu.u[k >> 1];
            unsigned bits = (k & 1) ? (word >> 16) : (word & 0xFFFFu);
            wk[k] = bf2f(bits);
        }

        const float bn = bias[i0 + n];
        const float en = (float)ex[i0 + n];

        // ---- 16-step triangular fixup (bpermute broadcast of new node vals) ----
        #pragma unroll
        for (int k = 0; k < 16; ++k) {
            if (n == k) {
                #pragma unroll
                for (int v = 0; v < 8; ++v)
                    val[v] = (sigmoidf(c[v]) + bn) * en;
            }
            const int srcIdx = (k + hi16) << 2;
            #pragma unroll
            for (int v = 0; v < 8; ++v) {
                int bv = __builtin_amdgcn_ds_bpermute(
                             srcIdx, __builtin_bit_cast(int, val[v]));
                c[v] += __builtin_bit_cast(float, bv) * wk[k];
            }
        }

        // ---- commit this block's node values into the LDS history tile ----
        {
            unsigned short* dcol = res + (INPUT_DIM + i0 + n);
            #pragma unroll
            for (int v = 0; v < 8; ++v)
                dcol[(size_t)(v + 8 * h) * ROWSTRIDE] = f2bf(val[v]);
        }
    }

    // ---- 4) last block (nodes 768..783) == the 16 outputs ----
    {
        float* obase = out + (size_t)rowBase * OUTPUT_DIM + n;
        #pragma unroll
        for (int v = 0; v < 8; ++v)
            obase[(size_t)(v + 8 * h) * OUTPUT_DIM] = val[v];
    }
}

// ---------------------------------------------------------------------------
extern "C" void kernel_launch(void* const* d_in, const int* in_sizes, int n_in,
                              void* d_out, int out_size, void* d_ws, size_t ws_size,
                              hipStream_t stream) {
    const float* x    = (const float*)d_in[0];
    const float* w    = (const float*)d_in[1];
    const float* bias = (const float*)d_in[2];
    const int*   conn = (const int*)d_in[3];
    const int*   ex   = (const int*)d_in[4];
    // d_in[5] = output_dim scalar (fixed at 16 here)

    unsigned short* effT = (unsigned short*)d_ws;   // 784*1040*2 B = 1.63 MB

    {
        int total  = NUM_MIDDLE * NUM_NODES;
        int blocks = (total + 255) / 256;
        eff_transpose_kernel<<<blocks, 256, 0, stream>>>(w, conn, effT);
    }
    {
        size_t lds_bytes = (size_t)WAVES_PER_BLOCK * ROWS_PER_WAVE * ROWSTRIDE * 2; // 268,288
        mlp_scan_kernel<<<BATCH / ROWS_PER_BLOCK, WAVES_PER_BLOCK * 32, lds_bytes, stream>>>(
            x, effT, bias, ex, (float*)d_out);
    }
}
